// SpectralConvWithManualFreqs_4947802325032
// MI455X (gfx1250) — matrix operand order repaired and checked by reading, not compiled, not run on hardware
//
#include <hip/hip_runtime.h>

// ---------------------------------------------------------------------------
// SpectralConv with Fibonacci mode selection, decomposed into 5 WMMA GEMMs.
// All operand tiles are stored "fragment-ready" (pre-swizzled to this file's WMMA
// lane layout, k = 16*(lane/16) + slot), so each fragment load is one contiguous 32B/lane load.
// LDS staging uses CDNA5 async-to-LDS (GLOBAL_LOAD_ASYNC_TO_LDS_B128).
// B=16, CIN=COUT=64, H=W=256, modes 32x32, norm='forward' (1/65536 at stage 5)
// ---------------------------------------------------------------------------

typedef __attribute__((ext_vector_type(16))) _Float16 v16h;
typedef __attribute__((ext_vector_type(8)))  _Float16 v8h;
typedef __attribute__((ext_vector_type(8)))  float    v8f;

__constant__ int IDX_H[32] = {0,1,2,3,5,8,13,21,34,55,89,144,233,
                              4,6,7,9,10,11,12,14,15,16,17,18,19,20,22,23,24,25,26};
__constant__ int IDX_W[32] = {0,1,2,3,5,8,13,21,34,55,89,
                              4,6,7,9,10,11,12,14,15,16,17,18,19,20,22,23,24,25,26,27,28};

// Flat position of element (k,n) inside a fragment-ready 32x16 B tile (512 f16)
__device__ __forceinline__ int bfrag_idx(int kl, int nl) {
  return ((kl >> 4) << 8) + (nl << 4) + (kl & 15);   // lane=(kl>>4)*16+nl, slot=kl&15
}
// Flat position of element (r,k) inside a fragment-ready 16x32 A tile (512 f16)
__device__ __forceinline__ int afrag_idx(int r, int kl) {
  return (((kl >> 4) << 4) + r) * 16 + (kl & 15);
}
__device__ __forceinline__ v16h frag_load(const _Float16* base) {
  return *(const v16h*)(base + (threadIdx.x & 31) * 16);   // 32B contiguous / lane
}
static __device__ __forceinline__ v8f wmma16(v16h a, v16h b, v8f c) {
  return __builtin_amdgcn_wmma_f32_16x16x32_f16(false, a, false, b, (short)0, c,
                                                false, false);
}

// CDNA5 async copy global->LDS, 16B/lane granules (ASYNCcnt tracked).
// Generic LDS addresses carry the LDS offset in bits[31:0] (ISA 10.2).
__device__ __forceinline__ void copy_lds_async(_Float16* dst, const _Float16* src,
                                               int n) {
  unsigned lbase = (unsigned)(uintptr_t)dst;
  for (int t = threadIdx.x; t < (n >> 3); t += blockDim.x) {
    unsigned loff = lbase + t * 16;
    const void* g = (const char*)src + (size_t)t * 16;
    asm volatile("global_load_async_to_lds_b128 %0, %1, off"
                 :: "v"(loff), "v"(g) : "memory");
  }
}
__device__ __forceinline__ void wait_async_lds() {
  asm volatile("s_wait_asynccnt 0x0" ::: "memory");
}

// A fragment straight from f32 global rows (stage 1 only)
__device__ __forceinline__ v16h load_a16_f32(const float* src, int ld) {
  int lane = threadIdx.x & 31;
  int r = lane & 15, half = lane >> 4;
  v16h a;
#pragma unroll
  for (int j = 0; j < 8; ++j) {
    int kb = 16 * half + 2 * j;
    a[2*j]   = (_Float16)src[r*ld + kb];
    a[2*j+1] = (_Float16)src[r*ld + kb + 1];
  }
  return a;
}

// ---- prep: fragment-ready twiddle tables (f16) ----------------------------
__global__ void k_prep_tw(_Float16* Fw, _Float16* EhC, _Float16* EhS, _Float16* EhSN,
                          _Float16* IEC, _Float16* IES, _Float16* IESN, _Float16* Bw) {
  int idx = blockIdx.x * blockDim.x + threadIdx.x;
  const float K256 = 6.28318530717958647692f / 256.0f;
  if (idx < 16384) {                       // Fw (w,n): n<32 cos, n>=32 -sin
    int w = idx >> 6, n = idx & 63;
    int kw = IDX_W[n & 31];
    float s, c; __sincosf(K256 * (float)((kw * w) & 255), &s, &c);
    int d = ((w >> 5) * 4 + (n >> 4)) * 512 + bfrag_idx(w & 31, n & 15);
    Fw[d] = (_Float16)(n < 32 ? c : -s);
  } else if (idx < 24576) {                // Eh (kh,h) A-tables
    int j = idx - 16384; int kh = j >> 8, h = j & 255;
    float s, c; __sincosf(K256 * (float)((IDX_H[kh] * h) & 255), &s, &c);
    int d = ((kh >> 4) * 8 + (h >> 5)) * 512 + afrag_idx(kh & 15, h & 31);
    EhC[d] = (_Float16)c; EhS[d] = (_Float16)s; EhSN[d] = (_Float16)(-s);
  } else if (idx < 32768) {                // IE (h,kh) A-tables
    int j = idx - 24576; int h = j >> 5, kh = j & 31;
    float s, c; __sincosf(K256 * (float)((IDX_H[kh] * h) & 255), &s, &c);
    int d = (h >> 4) * 512 + afrag_idx(h & 15, kh);
    IEC[d] = (_Float16)c; IES[d] = (_Float16)s; IESN[d] = (_Float16)(-s);
  } else if (idx < 49152) {                // Bw (kr,w) with Hermitian doubling
    int j = idx - 32768; int kr = j >> 8, w = j & 255;
    int kw = IDX_W[kr & 31];
    float cf = (kw == 0) ? 1.0f : 2.0f;
    float s, c; __sincosf(K256 * (float)((kw * w) & 255), &s, &c);
    int d = ((kr >> 5) * 16 + (w >> 4)) * 512 + bfrag_idx(kr & 31, w & 15);
    Bw[d] = (_Float16)(kr < 32 ? cf * c : -cf * s);
  }
}

// ---- prep: weights (CIN,COUT,32,32) f32 -> fragment-ready [mode][i][o] f16
__global__ void k_prep_w(const float* wr, const float* wi,
                         _Float16* WpR, _Float16* WpI) {
  int idx = blockIdx.x * blockDim.x + threadIdx.x;
  if (idx >= 1024 * 64 * 64) return;
  int mode = idx >> 12, rem = idx & 4095, i = rem >> 6, o = rem & 63;
  size_t src = (size_t)(i * 64 + o) * 1024 + mode;
  size_t d = (size_t)mode * 4096 + ((i >> 5) * 4 + (o >> 4)) * 512
           + bfrag_idx(i & 31, o & 15);
  WpR[d] = (_Float16)wr[src];
  WpI[d] = (_Float16)wi[src];
}

// ---- stage 1: W-DFT. Tf = fragment-ready [bid][ks4+ct][512] ---------------
__global__ void __launch_bounds__(128) k1_dftw(const float* x, const _Float16* Fwf,
                                               _Float16* Tf) {
  __shared__ __align__(32) _Float16 sFw[16384];               // 32 KB
  copy_lds_async(sFw, Fwf, 16384);
  int mtile = blockIdx.x * 4 + (threadIdx.x >> 5);
  const float* xrow = x + (size_t)mtile * 16 * 256;
  {  // warm L2/WGP$ with this wave's whole 16KB A-block (global_prefetch_b8)
    int lane = threadIdx.x & 31;
#pragma unroll
    for (int p = 0; p < 4; ++p)
      __builtin_prefetch(xrow + (lane & 15) * 256 + (lane >> 4) * 32 + p * 64, 0, 3);
  }
  wait_async_lds();
  __syncthreads();
  v8f acc[4] = {{}, {}, {}, {}};
#pragma unroll
  for (int ks = 0; ks < 8; ++ks) {
    v16h a = load_a16_f32(xrow + ks * 32, 256);
#pragma unroll
    for (int nt = 0; nt < 4; ++nt)
      acc[nt] = wmma16(a, frag_load(&sFw[(ks * 4 + nt) * 512]), acc[nt]);
  }
  int lane = threadIdx.x & 31, n = lane & 15, mh = (lane >> 4) * 8;
  int m0 = mtile * 16, bid = m0 >> 8, h0 = m0 & 255;
  int unitbase = (h0 >> 5) << 2, klhi = (h0 >> 4) & 1;
#pragma unroll
  for (int nt = 0; nt < 4; ++nt) {
    v8h tmp;
#pragma unroll
    for (int j = 0; j < 8; ++j) tmp[j] = (_Float16)acc[nt][j];
    // slot index = h&15 = mh+j -> contiguous 16B store per lane
    size_t base = (size_t)bid * 16384 + (size_t)(unitbase + nt) * 512
                + (klhi << 8) + (n << 4) + mh;
    *(v8h*)(Tf + base) = tmp;
  }
}

// ---- stage 2: H-DFT. Xr = C·Tr + S·Ti ; Xi = C·Ti - S·Tr ------------------
// Output: Xsel f32, fragment-ready A-order for stage 3: [mode][aoff 0..1023]
__global__ void __launch_bounds__(128) k2_dfth(const _Float16* Tf,
    const _Float16* EhCf, const _Float16* EhSf, const _Float16* EhSNf,
    float* XR, float* XI) {
  __shared__ __align__(32) _Float16 sC[8192], sS[8192], sSN[8192];  // 48 KB
  copy_lds_async(sC, EhCf, 8192);
  copy_lds_async(sS, EhSf, 8192);
  copy_lds_async(sSN, EhSNf, 8192);
  wait_async_lds();
  __syncthreads();
  int bid = blockIdx.x;                                // = b*CIN + i
  const _Float16* Tb = Tf + (size_t)bid * 16384;
  int wave = threadIdx.x >> 5;
  int isI = wave >> 1, mtile = wave & 1;
  const _Float16* sA2 = isI ? sSN : sS;
  int ct1 = isI ? 2 : 0, ct2 = isI ? 0 : 2;            // Tr cols=ct0..1, Ti=2..3
  v8f acc[2] = {{}, {}};
#pragma unroll
  for (int ks = 0; ks < 8; ++ks) {
    v16h a1 = frag_load(&sC[(mtile * 8 + ks) * 512]);
    v16h a2 = frag_load(&sA2[(mtile * 8 + ks) * 512]);
#pragma unroll
    for (int nt = 0; nt < 2; ++nt) {
      acc[nt] = wmma16(a1, frag_load(&Tb[(ks * 4 + ct1 + nt) * 512]), acc[nt]);
      acc[nt] = wmma16(a2, frag_load(&Tb[(ks * 4 + ct2 + nt) * 512]), acc[nt]);
    }
  }
  int lane = threadIdx.x & 31, n = lane & 15, mh = (lane >> 4) * 8;
  int b = bid >> 6, i = bid & 63;
  int aoff = ((i >> 5) << 9) + afrag_idx(b, i & 31);   // constant per block
  float* dst = isI ? XI : XR;
#pragma unroll
  for (int nt = 0; nt < 2; ++nt)
#pragma unroll
    for (int j = 0; j < 8; ++j) {
      int kh = mtile * 16 + mh + j, kw = nt * 16 + n;
      dst[(size_t)(kh * 32 + kw) * 1024 + aoff] = acc[nt][j];
    }
}

// ---- stage 3: per-mode channel mix. Or = Ar·Wr - Ai·Wi ; Oi = Ar·Wi + Ai·Wr
// Output: Osel f16, fragment-ready B-order for stage 4: [b*64+o][koff 0..1023]
__global__ void __launch_bounds__(128) k3_mix(const float* XR, const float* XI,
    const _Float16* WpR, const _Float16* WpI, _Float16* OR_, _Float16* OI_) {
  __shared__ __align__(32) _Float16 sAr[1024], sAi[1024], sAiN[1024];
  __shared__ __align__(32) _Float16 sWr[4096], sWi[4096];
  int mode = blockIdx.x;
  copy_lds_async(sWr, WpR + (size_t)mode * 4096, 4096);
  copy_lds_async(sWi, WpI + (size_t)mode * 4096, 4096);
  for (int t = threadIdx.x; t < 1024; t += 128) {      // straight copy (A-order)
    float r  = XR[(size_t)mode * 1024 + t];
    float im = XI[(size_t)mode * 1024 + t];
    sAr[t] = (_Float16)r; sAi[t] = (_Float16)im; sAiN[t] = (_Float16)(-im);
  }
  wait_async_lds();
  __syncthreads();
  int nt = threadIdx.x >> 5;                           // one o-tile per wave
  v8f cR = {}, cI = {};
#pragma unroll
  for (int ks = 0; ks < 2; ++ks) {
    v16h ar  = frag_load(&sAr[ks * 512]);
    v16h ai  = frag_load(&sAi[ks * 512]);
    v16h ain = frag_load(&sAiN[ks * 512]);
    v16h bwr = frag_load(&sWr[(ks * 4 + nt) * 512]);
    v16h bwi = frag_load(&sWi[(ks * 4 + nt) * 512]);
    cR = wmma16(ar, bwr, cR);  cR = wmma16(ain, bwi, cR);
    cI = wmma16(ar, bwi, cI);  cI = wmma16(ai,  bwr, cI);
  }
  int lane = threadIdx.x & 31, n = lane & 15, mh = (lane >> 4) * 8;
  int kh = mode >> 5, kw = mode & 31;
  int koff = ((kw >> 4) << 9) + bfrag_idx(kh, kw & 15);  // constant per block
#pragma unroll
  for (int j = 0; j < 8; ++j) {
    size_t d = (size_t)((mh + j) * 64 + nt * 16 + n) * 1024 + koff;
    OR_[d] = (_Float16)cR[j];
    OI_[d] = (_Float16)cI[j];
  }
}

// ---- stage 4: inverse H. Ur = C·Sr - S·Si ; Ui = C·Si + S·Sr --------------
// Output: Uf f16, fragment-ready A-order for stage 5: [rowtile][ks*512+aoff]
__global__ void __launch_bounds__(128) k4_idfth(const _Float16* ORf, const _Float16* OIf,
    const _Float16* IECf, const _Float16* IESf, const _Float16* IESNf, _Float16* Uf) {
  __shared__ __align__(32) _Float16 sC[8192], sS[8192], sSN[8192];  // 48 KB
  __shared__ __align__(32) _Float16 sSr[1024], sSi[1024];
  int bo = blockIdx.x;
  copy_lds_async(sC, IECf, 8192);
  copy_lds_async(sS, IESf, 8192);
  copy_lds_async(sSN, IESNf, 8192);
  copy_lds_async(sSr, ORf + (size_t)bo * 1024, 1024);
  copy_lds_async(sSi, OIf + (size_t)bo * 1024, 1024);
  wait_async_lds();
  __syncthreads();
  int wave = threadIdx.x >> 5;
  v16h bSr0 = frag_load(&sSr[0]), bSr1 = frag_load(&sSr[512]);
  v16h bSi0 = frag_load(&sSi[0]), bSi1 = frag_load(&sSi[512]);
  int lane = threadIdx.x & 31, n = lane & 15, mh = (lane >> 4) * 8;
#pragma unroll
  for (int mt = 0; mt < 4; ++mt) {
    int mtile = wave * 4 + mt;                          // 0..15 h-tiles
    v16h aC  = frag_load(&sC[mtile * 512]);
    v16h aS  = frag_load(&sS[mtile * 512]);
    v16h aSN = frag_load(&sSN[mtile * 512]);
    v8f cR0 = {}, cR1 = {}, cI0 = {}, cI1 = {};
    cR0 = wmma16(aC, bSr0, cR0); cR0 = wmma16(aSN, bSi0, cR0);
    cR1 = wmma16(aC, bSr1, cR1); cR1 = wmma16(aSN, bSi1, cR1);
    cI0 = wmma16(aC, bSi0, cI0); cI0 = wmma16(aS,  bSr0, cI0);
    cI1 = wmma16(aC, bSi1, cI1); cI1 = wmma16(aS,  bSr1, cI1);
    _Float16* Ub = Uf + ((size_t)bo * 16 + mtile) * 1024;
#pragma unroll
    for (int j = 0; j < 8; ++j) {
      Ub[      afrag_idx(mh + j, n)]      = (_Float16)cR0[j];
      Ub[      afrag_idx(mh + j, 16 + n)] = (_Float16)cR1[j];
      Ub[512 + afrag_idx(mh + j, n)]      = (_Float16)cI0[j];
      Ub[512 + afrag_idx(mh + j, 16 + n)] = (_Float16)cI1[j];
    }
  }
}

// ---- stage 5: inverse W (+1/65536 norm, +bias) ----------------------------
__global__ void __launch_bounds__(128) k5_idftw(const _Float16* Uf, const _Float16* Bwf,
                                                const float* bias, float* out) {
  __shared__ __align__(32) _Float16 sB[16384];               // 32 KB
  copy_lds_async(sB, Bwf, 16384);
  int mt = blockIdx.x * 4 + (threadIdx.x >> 5);              // row tile (b,o,h)/16
  __builtin_prefetch(Uf + (size_t)mt * 1024 + (threadIdx.x & 31) * 32, 0, 3);
  wait_async_lds();
  __syncthreads();
  v16h a0 = frag_load(Uf + (size_t)mt * 1024);
  v16h a1 = frag_load(Uf + (size_t)mt * 1024 + 512);
  float bv = bias[(mt >> 4) & 63];
  int lane = threadIdx.x & 31, n = lane & 15, mh = (lane >> 4) * 8;
  float* orow = out + (size_t)mt * 16 * 256;
  const float inv = 1.0f / 65536.0f;
#pragma unroll
  for (int nt = 0; nt < 16; ++nt) {
    v8f c = {};
    c = wmma16(a0, frag_load(&sB[nt * 512]), c);
    c = wmma16(a1, frag_load(&sB[(16 + nt) * 512]), c);
#pragma unroll
    for (int j = 0; j < 8; ++j)
      orow[(mh + j) * 256 + nt * 16 + n] = c[j] * inv + bv;
  }
}

// ---------------------------------------------------------------------------
extern "C" void kernel_launch(void* const* d_in, const int* in_sizes, int n_in,
                              void* d_out, int out_size, void* d_ws, size_t ws_size,
                              hipStream_t stream) {
  const float* x    = (const float*)d_in[0];   // (16,64,256,256)
  const float* wr   = (const float*)d_in[1];   // (64,64,32,32)
  const float* wi   = (const float*)d_in[2];   // (64,64,32,32)
  const float* bias = (const float*)d_in[3];   // (64,1,1)
  float* out = (float*)d_out;

  char* ws = (char*)d_ws;
  _Float16* Fw    = (_Float16*)(ws + 0);          //  32768 B
  _Float16* EhC   = (_Float16*)(ws + 32768);      //  16384
  _Float16* EhS   = (_Float16*)(ws + 49152);      //  16384
  _Float16* EhSN  = (_Float16*)(ws + 65536);      //  16384
  _Float16* IEC   = (_Float16*)(ws + 81920);      //  16384
  _Float16* IES   = (_Float16*)(ws + 98304);      //  16384
  _Float16* IESN  = (_Float16*)(ws + 114688);     //  16384
  _Float16* Bw    = (_Float16*)(ws + 131072);     //  32768
  _Float16* WpR   = (_Float16*)(ws + 163840);     //  8388608
  _Float16* WpI   = (_Float16*)(ws + 8552448);    //  8388608
  _Float16* Tf    = (_Float16*)(ws + 16941056);   // 33554432
  float*    XR    = (float*)   (ws + 50495488);   //  4194304
  float*    XI    = (float*)   (ws + 54689792);   //  4194304
  _Float16* ORf   = (_Float16*)(ws + 58884096);   //  2097152
  _Float16* OIf   = (_Float16*)(ws + 60981248);   //  2097152
  _Float16* Uf    = (_Float16*)(ws + 63078400);   // 33554432  (total ~92 MB)

  k_prep_tw<<<dim3(192), dim3(256), 0, stream>>>(Fw, EhC, EhS, EhSN,
                                                 IEC, IES, IESN, Bw);
  k_prep_w<<<dim3(16384), dim3(256), 0, stream>>>(wr, wi, WpR, WpI);
  k1_dftw<<<dim3(4096), dim3(128), 0, stream>>>(x, Fw, Tf);
  k2_dfth<<<dim3(1024), dim3(128), 0, stream>>>(Tf, EhC, EhS, EhSN, XR, XI);
  k3_mix <<<dim3(1024), dim3(128), 0, stream>>>(XR, XI, WpR, WpI, ORf, OIf);
  k4_idfth<<<dim3(1024), dim3(128), 0, stream>>>(ORf, OIf, IEC, IES, IESN, Uf);
  k5_idftw<<<dim3(4096), dim3(128), 0, stream>>>(Uf, Bw, bias, out);
}
